// LSTM_19834158973320
// MI455X (gfx1250) — matrix operand order repaired and checked by reading, not compiled, not run on hardware
//
#include <hip/hip_runtime.h>
#include <hip/hip_bf16.h>

// ---------------- problem constants ----------------
#define BB   64      // batch
#define TT   2048    // time steps
#define HH   1024    // hidden
#define DIN  68      // input features
#define DINP 96      // input features padded to multiple of 32 (K dim)
#define G4   4096    // 4*H gate columns
#define DOUT 64      // output features
#define NWG  64      // persistent workgroups
#define CPW  16      // hidden columns owned per workgroup (H / NWG)
#define NTHR 256     // 8 wave32 waves: wave g and g+4 split K for gate g

typedef __attribute__((ext_vector_type(16))) __bf16 v16bf;
typedef __attribute__((ext_vector_type(8)))  __bf16 v8bf;
typedef __attribute__((ext_vector_type(8)))  float  v8f;

// ---------------------------------------------------------------------------
// fragment helpers (layouts per CDNA5 ISA 7.12.2, wave32)
// ---------------------------------------------------------------------------
static __device__ __forceinline__ v16bf cat16(v8bf lo, v8bf hi) {
  v16bf r;
#pragma unroll
  for (int i = 0; i < 8; ++i) { r[i] = lo[i]; r[i + 8] = hi[i]; }
  return r;
}

// A fragment: 16x32 bf16 (MxK). Row-major source, rows m0..m0+15, cols ks..ks+31.
static __device__ __forceinline__ v16bf load_a(const __bf16* __restrict__ base,
                                               int stride, int m0, int ks, int lane) {
  const int m  = m0 + (lane & 15);
  const int kb = ks + ((lane >> 4) << 3);
  const __bf16* p = base + m * stride + kb;
  v8bf lo = *(const v8bf*)(p);        // global_load_b128
  v8bf hi = *(const v8bf*)(p + 16);   // global_load_b128
  return cat16(lo, hi);
}

// B fragment: 32x16 bf16 (KxN). Source is K-major [K][ncols] (pre-transposed).
static __device__ __forceinline__ v16bf load_b(const __bf16* __restrict__ base,
                                               int ncols, int n0, int ks, int lane) {
  const __bf16* p = base + (ks + ((lane >> 4) << 3)) * ncols + n0 + (lane & 15);
  v16bf r;
#pragma unroll
  for (int i = 0; i < 8; ++i) {
    r[i]     = p[i * ncols];
    r[i + 8] = p[(16 + i) * ncols];
  }
  return r;
}

static __device__ __forceinline__ v8f wmma_bf16(v16bf a, v16bf b, v8f c) {
  return __builtin_amdgcn_wmma_f32_16x16x32_bf16(false, a, false, b, (short)0, c,
                                                 false, false);
}

static __device__ __forceinline__ float sigf(float x) {
  return 1.0f / (1.0f + __expf(-x));
}

// device-wide barrier: monotonic counter in L2, reset by host memset each launch
static __device__ __forceinline__ void grid_barrier(unsigned* bar, unsigned tgt) {
  __threadfence();
  __syncthreads();
  if (threadIdx.x == 0) {
    __hip_atomic_fetch_add(bar, 1u, __ATOMIC_ACQ_REL, __HIP_MEMORY_SCOPE_AGENT);
    while (__hip_atomic_load(bar, __ATOMIC_ACQUIRE, __HIP_MEMORY_SCOPE_AGENT) < tgt)
      __builtin_amdgcn_s_sleep(2);
  }
  __syncthreads();
  __threadfence();
}

// ---------------------------------------------------------------------------
// software-pipelined K loop: one 16-wide N strip, 4 M-tiles (batch 0..63).
// Ping-pong fragment buffers (fa/fb): the loads for kstep i+1 are pinned
// above the 4 WMMAs of kstep i with sched_barrier so they stay in flight.
// ---------------------------------------------------------------------------
struct Frags { v16bf a0, a1, a2, a3, b; };

static __device__ __forceinline__ void load_frags(Frags& f,
                                                  const __bf16* __restrict__ A, int astride,
                                                  const __bf16* __restrict__ Bt, int bcols,
                                                  int n0, int ks, int lane) {
  f.b  = load_b(Bt, bcols, n0, ks, lane);
  f.a0 = load_a(A, astride,  0, ks, lane);
  f.a1 = load_a(A, astride, 16, ks, lane);
  f.a2 = load_a(A, astride, 32, ks, lane);
  f.a3 = load_a(A, astride, 48, ks, lane);
}

static __device__ __forceinline__ void do_wmma(const Frags& f, v8f acc[4]) {
  acc[0] = wmma_bf16(f.a0, f.b, acc[0]);
  acc[1] = wmma_bf16(f.a1, f.b, acc[1]);
  acc[2] = wmma_bf16(f.a2, f.b, acc[2]);
  acc[3] = wmma_bf16(f.a3, f.b, acc[3]);
}

static __device__ __forceinline__ void gemm_acc(const __bf16* __restrict__ A, int astride,
                                                int k0, int k1,
                                                const __bf16* __restrict__ Bt, int bcols,
                                                int n0, int lane, v8f acc[4]) {
  Frags fa, fb;
  int nk = (k1 - k0) >> 5;        // # of 32-wide k-steps (always >= 1 here)
  int ks = k0;
  load_frags(fa, A, astride, Bt, bcols, n0, ks, lane);
  ks += 32;
  int rem = nk - 1;
  while (rem >= 2) {
    load_frags(fb, A, astride, Bt, bcols, n0, ks, lane);       // prefetch i+1
    __builtin_amdgcn_sched_barrier(0);                         // keep loads up here
    do_wmma(fa, acc);
    load_frags(fa, A, astride, Bt, bcols, n0, ks + 32, lane);  // prefetch i+2
    __builtin_amdgcn_sched_barrier(0);
    do_wmma(fb, acc);
    ks += 64;
    rem -= 2;
  }
  if (rem == 1) {
    load_frags(fb, A, astride, Bt, bcols, n0, ks, lane);
    __builtin_amdgcn_sched_barrier(0);
    do_wmma(fa, acc);
    do_wmma(fb, acc);
  } else {
    do_wmma(fa, acc);
  }
}

// ---------------------------------------------------------------------------
// prep kernels (run once per launch; weights stay L2-resident afterwards)
// ---------------------------------------------------------------------------
__global__ void prep_bias_k(const float* __restrict__ bi0, const float* __restrict__ bh0,
                            const float* __restrict__ bi1, const float* __restrict__ bh1,
                            float* __restrict__ s0, float* __restrict__ s1) {
  int i = blockIdx.x * blockDim.x + threadIdx.x;
  if (i < G4) s0[i] = bi0[i] + bh0[i];
  else        s1[i - G4] = bi1[i - G4] + bh1[i - G4];
}

// fp32 W[N][K] (row-major) -> bf16 Wt[Kpad][N] (K-major), zero padded K rows
__global__ void prep_wt_k(const float* __restrict__ W, __bf16* __restrict__ Wt,
                          int N, int K) {
  int i = blockIdx.x * blockDim.x + threadIdx.x;  // i = k*N + n, grid covers Kpad*N
  int n = i % N, k = i / N;
  float v = (k < K) ? W[n * K + k] : 0.0f;
  Wt[i] = (__bf16)v;
}

// x[B][T][DIN] fp32 -> xpad[T][B][DINP] bf16 (zero padded)
__global__ void prep_x_k(const float* __restrict__ x, __bf16* __restrict__ xp) {
  int i = blockIdx.x * blockDim.x + threadIdx.x;  // i = (t*BB+b)*DINP + kp
  int kp = i % DINP;
  int bt = i / DINP;
  int b = bt % BB, t = bt / BB;
  float v = (kp < DIN) ? x[(b * TT + t) * DIN + kp] : 0.0f;
  xp[i] = (__bf16)v;
}

// ---------------------------------------------------------------------------
// persistent LSTM kernel: 64 WGs x 256 threads = 8 wave32 waves (2 per SIMD).
// WG owns CPW=16 hidden columns. Waves g and g+4 both compute gate g,
// splitting the K range; partials combined in LDS (store + accumulate phase).
// ---------------------------------------------------------------------------
__global__ __launch_bounds__(NTHR)
__attribute__((amdgpu_waves_per_eu(2, 2)))
void lstm_persistent(const __bf16* __restrict__ xpad,
                     const __bf16* __restrict__ Wih0t,
                     const __bf16* __restrict__ Whh0t,
                     const __bf16* __restrict__ Wih1t,
                     const __bf16* __restrict__ Whh1t,
                     const __bf16* __restrict__ Woutt,
                     const float*  __restrict__ bsum0,
                     const float*  __restrict__ bsum1,
                     const float*  __restrict__ bout,
                     __bf16* __restrict__ h0buf,   // [2][BB][HH]
                     __bf16* __restrict__ h1buf,   // [2][BB][HH]
                     unsigned* __restrict__ bar,
                     float* __restrict__ out) {    // [BB][TT][DOUT]
  __shared__ float gbuf[4][BB][CPW];               // 16 KB gate staging

  const int wg   = blockIdx.x;
  const int tid  = threadIdx.x;
  const int wave = tid >> 5;                       // 0..7
  const int lane = tid & 31;
  const int gate = wave & 3;
  const int half = wave >> 2;                      // 0: low-K half, 1: high-K half
  const int j0   = wg * CPW;                       // hidden-col slice base
  const int ng   = gate * HH + j0;                 // gate-col base in [0,4096)

  // per-thread cell state (4 elements each layer), fixed mapping across t
  float c0s[4] = {0.f, 0.f, 0.f, 0.f};
  float c1s[4] = {0.f, 0.f, 0.f, 0.f};

  // hoisted per-thread elementwise constants
  float bi0[4], bf0[4], bg0[4], bo0[4];
  float bi1[4], bf1[4], bg1[4], bo1[4];
  int   eb[4], ej[4], hidx[4];
#pragma unroll
  for (int k = 0; k < 4; ++k) {
    int e = tid + NTHR * k;
    eb[k] = e >> 4;
    ej[k] = e & 15;
    int c = j0 + ej[k];
    hidx[k] = eb[k] * HH + c;
    bi0[k] = bsum0[c];           bf0[k] = bsum0[HH + c];
    bg0[k] = bsum0[2 * HH + c];  bo0[k] = bsum0[3 * HH + c];
    bi1[k] = bsum1[c];           bf1[k] = bsum1[HH + c];
    bg1[k] = bsum1[2 * HH + c];  bo1[k] = bsum1[3 * HH + c];
  }

  // hoisted output-projection lane constants (used by wave0 of wg<16)
  const int omt = wg & 3, odt = (wg >> 2) & 3;
  const int od  = odt * 16 + (lane & 15);
  const int orb = omt * 16 + ((lane >> 4) << 3);
  const float obias = bout[od];

  // staging coordinates
  const int scol = lane & 15;
  const int srb  = (lane >> 4) << 3;

  unsigned bar_tgt = 0;

  for (int t = 0; t < TT; ++t) {
    const int cur = t & 1, nxt = cur ^ 1;
    const __bf16* h0c = h0buf + cur * (BB * HH);
    const __bf16* h1c = h1buf + cur * (BB * HH);
    __bf16* h0n = h0buf + nxt * (BB * HH);
    __bf16* h1n = h1buf + nxt * (BB * HH);

    // ================= layer 0: gates = x_t @ Wih0^T + h0 @ Whh0^T ==========
    {
      v8f acc[4] = {};
      if (half == 0) {
        gemm_acc(xpad + t * (BB * DINP), DINP, 0, DINP, Wih0t, G4, ng, lane, acc);
        gemm_acc(h0c, HH, 0, HH / 2, Whh0t, G4, ng, lane, acc);
      } else {
        gemm_acc(h0c, HH, HH / 2, HH, Whh0t, G4, ng, lane, acc);
      }
      if (half == 0) {                 // phase A: store partials
#pragma unroll
        for (int r = 0; r < 8; ++r) {
          gbuf[gate][ 0 + srb + r][scol] = acc[0][r];
          gbuf[gate][16 + srb + r][scol] = acc[1][r];
          gbuf[gate][32 + srb + r][scol] = acc[2][r];
          gbuf[gate][48 + srb + r][scol] = acc[3][r];
        }
      }
      __syncthreads();
      if (half == 1) {                 // phase B: accumulate partials
#pragma unroll
        for (int r = 0; r < 8; ++r) {
          gbuf[gate][ 0 + srb + r][scol] += acc[0][r];
          gbuf[gate][16 + srb + r][scol] += acc[1][r];
          gbuf[gate][32 + srb + r][scol] += acc[2][r];
          gbuf[gate][48 + srb + r][scol] += acc[3][r];
        }
      }
    }
    __syncthreads();
#pragma unroll
    for (int k = 0; k < 4; ++k) {
      float gi = sigf (gbuf[0][eb[k]][ej[k]] + bi0[k]);
      float gf = sigf (gbuf[1][eb[k]][ej[k]] + bf0[k]);
      float gg = tanhf(gbuf[2][eb[k]][ej[k]] + bg0[k]);
      float go = sigf (gbuf[3][eb[k]][ej[k]] + bo0[k]);
      float c  = gf * c0s[k] + gi * gg;
      c0s[k]   = c;
      h0n[hidx[k]] = (__bf16)(go * tanhf(c));
    }
    bar_tgt += NWG;
    grid_barrier(bar, bar_tgt);   // h0_new visible device-wide

    // ================= layer 1: gates = h0_new @ Wih1^T + h1 @ Whh1^T =======
    {
      v8f acc[4] = {};
      if (half == 0) {
        gemm_acc(h0n, HH, 0, HH, Wih1t, G4, ng, lane, acc);
      } else {
        gemm_acc(h1c, HH, 0, HH, Whh1t, G4, ng, lane, acc);
      }
      if (half == 0) {
#pragma unroll
        for (int r = 0; r < 8; ++r) {
          gbuf[gate][ 0 + srb + r][scol] = acc[0][r];
          gbuf[gate][16 + srb + r][scol] = acc[1][r];
          gbuf[gate][32 + srb + r][scol] = acc[2][r];
          gbuf[gate][48 + srb + r][scol] = acc[3][r];
        }
      }
      __syncthreads();
      if (half == 1) {
#pragma unroll
        for (int r = 0; r < 8; ++r) {
          gbuf[gate][ 0 + srb + r][scol] += acc[0][r];
          gbuf[gate][16 + srb + r][scol] += acc[1][r];
          gbuf[gate][32 + srb + r][scol] += acc[2][r];
          gbuf[gate][48 + srb + r][scol] += acc[3][r];
        }
      }
    }
    __syncthreads();
#pragma unroll
    for (int k = 0; k < 4; ++k) {
      float gi = sigf (gbuf[0][eb[k]][ej[k]] + bi1[k]);
      float gf = sigf (gbuf[1][eb[k]][ej[k]] + bf1[k]);
      float gg = tanhf(gbuf[2][eb[k]][ej[k]] + bg1[k]);
      float go = sigf (gbuf[3][eb[k]][ej[k]] + bo1[k]);
      float c  = gf * c1s[k] + gi * gg;
      c1s[k]   = c;
      h1n[hidx[k]] = (__bf16)(go * tanhf(c));
    }
    bar_tgt += NWG;
    grid_barrier(bar, bar_tgt);   // h1_new visible device-wide

    // ================= output projection: out_t = h1_new @ Wout^T + b =======
    // 16 WGs each compute one 16x16 tile (wave 0 only). Safe to overlap with
    // the next step: the writer of this h1 buffer (step t+2) is gated by
    // barriers this WG must also reach.
    if (wg < 16 && wave == 0) {
      v8f acc = {};
      for (int ks = 0; ks < HH; ks += 32) {
        v16bf bf = load_b(Woutt, DOUT, odt * 16, ks, lane);
        v16bf af = load_a(h1n, HH, omt * 16, ks, lane);
        acc = wmma_bf16(af, bf, acc);
      }
#pragma unroll
      for (int r = 0; r < 8; ++r) {
        int b = orb + r;
        out[(b * TT + t) * DOUT + od] = acc[r] + obias;
      }
    }
  }
}

// ---------------------------------------------------------------------------
// host launcher
// ---------------------------------------------------------------------------
extern "C" void kernel_launch(void* const* d_in, const int* in_sizes, int n_in,
                              void* d_out, int out_size, void* d_ws, size_t ws_size,
                              hipStream_t stream) {
  (void)in_sizes; (void)n_in; (void)out_size; (void)ws_size;

  const float* x     = (const float*)d_in[0];
  const float* W_ih0 = (const float*)d_in[1];
  const float* W_hh0 = (const float*)d_in[2];
  const float* b_ih0 = (const float*)d_in[3];
  const float* b_hh0 = (const float*)d_in[4];
  const float* W_ih1 = (const float*)d_in[5];
  const float* W_hh1 = (const float*)d_in[6];
  const float* b_ih1 = (const float*)d_in[7];
  const float* b_hh1 = (const float*)d_in[8];
  const float* W_out = (const float*)d_in[9];
  const float* b_out = (const float*)d_in[10];
  float* out = (float*)d_out;

  // workspace carve-up (~52 MB)
  char* ws = (char*)d_ws;
  size_t off = 0;
  auto carve = [&](size_t bytes) -> void* {
    void* p = ws + off;
    off = (off + bytes + 255) & ~(size_t)255;
    return p;
  };
  unsigned* bar   = (unsigned*)carve(256);
  __bf16*   h0buf = (__bf16*)carve((size_t)2 * BB * HH * 2);
  __bf16*   h1buf = (__bf16*)carve((size_t)2 * BB * HH * 2);
  float*    bsum0 = (float*)carve((size_t)G4 * 4);
  float*    bsum1 = (float*)carve((size_t)G4 * 4);
  __bf16*   Wih0t = (__bf16*)carve((size_t)DINP * G4 * 2);
  __bf16*   Whh0t = (__bf16*)carve((size_t)HH * G4 * 2);
  __bf16*   Wih1t = (__bf16*)carve((size_t)HH * G4 * 2);
  __bf16*   Whh1t = (__bf16*)carve((size_t)HH * G4 * 2);
  __bf16*   Woutt = (__bf16*)carve((size_t)HH * DOUT * 2);
  __bf16*   xpad  = (__bf16*)carve((size_t)TT * BB * DINP * 2);

  // reset barrier + zero initial hidden state every launch (graph-safe)
  hipMemsetAsync((void*)bar, 0, 256, stream);
  hipMemsetAsync((void*)h0buf, 0, (size_t)2 * BB * HH * 2, stream);
  hipMemsetAsync((void*)h1buf, 0, (size_t)2 * BB * HH * 2, stream);

  // weight / bias / input preprocessing (fp32 -> bf16, K-major transpose)
  prep_bias_k<<<(2 * G4) / 256, 256, 0, stream>>>(b_ih0, b_hh0, b_ih1, b_hh1,
                                                  bsum0, bsum1);
  prep_wt_k<<<(DINP * G4) / 256, 256, 0, stream>>>(W_ih0, Wih0t, G4, DIN);
  prep_wt_k<<<(HH * G4) / 256, 256, 0, stream>>>(W_hh0, Whh0t, G4, HH);
  prep_wt_k<<<(HH * G4) / 256, 256, 0, stream>>>(W_ih1, Wih1t, G4, HH);
  prep_wt_k<<<(HH * G4) / 256, 256, 0, stream>>>(W_hh1, Whh1t, G4, HH);
  prep_wt_k<<<(HH * DOUT) / 256, 256, 0, stream>>>(W_out, Woutt, DOUT, HH);
  prep_x_k<<<((size_t)TT * BB * DINP) / 256, 256, 0, stream>>>(x, xpad);

  // persistent recurrent kernel: whole time loop on-device
  lstm_persistent<<<NWG, NTHR, 0, stream>>>(xpad, Wih0t, Whh0t, Wih1t, Whh1t,
                                            Woutt, bsum0, bsum1, b_out,
                                            h0buf, h1buf, bar, out);
}
